// FlashAttention_77008763617648
// MI455X (gfx1250) — matrix operand order repaired and checked by reading, not compiled, not run on hardware
//
#include <hip/hip_runtime.h>

typedef __attribute__((ext_vector_type(16))) _Float16 v16h;
typedef __attribute__((ext_vector_type(8)))  float    v8f;
typedef __attribute__((ext_vector_type(4)))  float    v4f;
typedef __attribute__((ext_vector_type(4)))  int      v4i;

#define B_  4
#define L_  2048
#define H_  16
#define D_  64
#define KC  32                    // keys per chunk
#define NCHUNK (L_ / KC)
#define NWAVES 4
#define QT  2                     // query tiles (of 16) per wave -> 128 q per block

#define HAS_ASYNC_LDS  __has_builtin(__builtin_amdgcn_global_load_async_to_lds_b128)

#if HAS_ASYNC_LDS
typedef __attribute__((address_space(1))) v4i as1_v4i;   // global int4*
typedef __attribute__((address_space(3))) v4i as3_v4i;   // LDS int4*
#endif

__device__ __forceinline__ void wait_async_zero() {
#if HAS_ASYNC_LDS
#if __has_builtin(__builtin_amdgcn_s_wait_asynccnt)
    __builtin_amdgcn_s_wait_asynccnt(0);
#else
    asm volatile("s_wait_asynccnt 0" ::: "memory");
#endif
#endif
}

__global__ __launch_bounds__(128) void fa_fwd_kernel(const float* __restrict__ Q,
                                                     const float* __restrict__ Kp,
                                                     const float* __restrict__ Vp,
                                                     float* __restrict__ O)
{
    // Raw fp32 chunks, double buffered. K: [key][d], V: [key][d]
    __shared__ __align__(16) float Kf[2][KC * D_];   // 2 x 8 KB
    __shared__ __align__(16) float Vf[2][KC * D_];   // 2 x 8 KB

    const int tid  = threadIdx.x;
    const int lane = tid & 31;
    const int wave = tid >> 5;
    const int qb = blockIdx.x;   // 0..15 (query block of 128)
    const int h  = blockIdx.y;   // head
    const int b  = blockIdx.z;   // batch

    const int hf  = lane >> 4;   // half-wave: 0 or 1
    const int l15 = lane & 15;

    const int qbase = qb * (64 * QT) + wave * (16 * QT);  // wave's first query

    // ---- stage one 32-key chunk of K and V into LDS buffer `buf` ----
    auto stage = [&](int kcIdx, int buf) {
        const int key0 = kcIdx * KC;
        for (int v = 0; v < 4; ++v) {
            const int idx4 = v * 128 + tid;          // float4 index into 32x64
            const int lin  = idx4 * 4;
            const int key  = lin >> 6;
            const int d    = lin & 63;
            const size_t g = (((size_t)b * L_ + key0 + key) * H_ + h) * D_ + d;
#if HAS_ASYNC_LDS
            __builtin_amdgcn_global_load_async_to_lds_b128(
                (as1_v4i*)(Kp + g), (as3_v4i*)&Kf[buf][lin], 0, 0);
            __builtin_amdgcn_global_load_async_to_lds_b128(
                (as1_v4i*)(Vp + g), (as3_v4i*)&Vf[buf][lin], 0, 0);
#else
            *reinterpret_cast<v4f*>(&Kf[buf][lin]) = *reinterpret_cast<const v4f*>(Kp + g);
            *reinterpret_cast<v4f*>(&Vf[buf][lin]) = *reinterpret_cast<const v4f*>(Vp + g);
#endif
        }
    };

    // ---- Q^T B-fragments (32d x 16q) with 1/sqrt(d) folded in, kept in regs ----
    // B layout: lane holds column N = l15; element j -> contraction K = j + 16*hf (+32*kk)
    v16h bq[QT][2];
    for (int qt = 0; qt < QT; ++qt) {
        const int qg = qbase + qt * 16 + l15;
        const float* qp = Q + (((size_t)b * L_ + qg) * H_ + h) * D_;
        for (int kk = 0; kk < 2; ++kk) {
            const int d0 = 16 * hf + 32 * kk;
            const v4f* p = reinterpret_cast<const v4f*>(qp + d0);
            for (int v = 0; v < 4; ++v) {
                v4f f = p[v];
                for (int i = 0; i < 4; ++i)
                    bq[qt][kk][v * 4 + i] = (_Float16)(f[i] * 0.125f);
            }
        }
    }

    float m_run[QT], l_run[QT];
    v8f o[QT][4];                  // o[qt][t][r] = O[q=l15][d = 16t + 8*hf + r]
    for (int qt = 0; qt < QT; ++qt) {
        m_run[qt] = -1e30f; l_run[qt] = 0.0f;
        for (int t = 0; t < 4; ++t) o[qt][t] = (v8f){};
    }

    stage(0, 0);           // prologue: fill buffer 0

    for (int kc = 0; kc < NCHUNK; ++kc) {
        const int cur = kc & 1;
        wait_async_zero();     // my async deposits (for buf cur) have landed in LDS
        __syncthreads();       // everyone's deposits visible; prev compute done

        if (kc + 1 < NCHUNK) stage(kc + 1, cur ^ 1);   // overlap DMA with compute

        // ---- S^T = (K/8) * Q^T : per q-tile, two 16x16 tiles (keys 0..15/16..31)
        // K A-fragments are shared by both q-tiles.
        v8f s[QT][2];
        for (int qt = 0; qt < QT; ++qt) { s[qt][0] = (v8f){}; s[qt][1] = (v8f){}; }
        for (int kk = 0; kk < 2; ++kk) {
            // A layout: lane row m = l15; element j -> K(d) = j + 16*hf (+32*kk)
            const int d0 = 16 * hf + 32 * kk;
            v16h a0, a1;
            {
                const float* p0 = &Kf[cur][l15 * D_ + d0];
                const float* p1 = &Kf[cur][(16 + l15) * D_ + d0];
                for (int i = 0; i < 8; ++i) {
                    a0[i]     = (_Float16)p0[i];
                    a0[8 + i] = (_Float16)p0[8 + i];
                    a1[i]     = (_Float16)p1[i];
                    a1[8 + i] = (_Float16)p1[8 + i];
                }
            }
            for (int qt = 0; qt < QT; ++qt) {
                s[qt][0] = __builtin_amdgcn_wmma_f32_16x16x32_f16(false, a0, false, bq[qt][kk],
                                                                  (short)0, s[qt][0], false, false);
                s[qt][1] = __builtin_amdgcn_wmma_f32_16x16x32_f16(false, a1, false, bq[qt][kk],
                                                                  (short)0, s[qt][1], false, false);
            }
        }

        // ---- online softmax per q-tile + P^T B-fragment build ----
        v16h bp[QT];
        for (int qt = 0; qt < QT; ++qt) {
            v8f s0 = s[qt][0], s1 = s[qt][1];
            float mloc = s0[0];
            for (int r = 1; r < 8; ++r) mloc = fmaxf(mloc, s0[r]);
            for (int r = 0; r < 8; ++r) mloc = fmaxf(mloc, s1[r]);
            mloc = fmaxf(mloc, __shfl_xor(mloc, 16, 32));
            const float m_new = fmaxf(m_run[qt], mloc);
            const float alpha = __expf(m_run[qt] - m_new);
            float p0v[8], p1v[8], sloc = 0.0f;
            for (int r = 0; r < 8; ++r) {
                p0v[r] = __expf(s0[r] - m_new);
                p1v[r] = __expf(s1[r] - m_new);
                sloc += p0v[r] + p1v[r];
            }
            sloc += __shfl_xor(sloc, 16, 32);
            l_run[qt] = l_run[qt] * alpha + sloc;
            m_run[qt] = m_new;
            for (int t = 0; t < 4; ++t)
                for (int r = 0; r < 8; ++r) o[qt][t][r] *= alpha;

            // half 0 lane needs keys 0..15  : j=r local(p0), j=8+r from partner's p0
            // half 1 lane needs keys 16..31 : j=r from partner's p1, j=8+r local(p1)
            for (int r = 0; r < 8; ++r) {
                const float sw0 = __shfl_xor(p0v[r], 16, 32);
                const float sw1 = __shfl_xor(p1v[r], 16, 32);
                bp[qt][r]     = (_Float16)(hf ? sw1 : p0v[r]);
                bp[qt][8 + r] = (_Float16)(hf ? p1v[r] : sw0);
            }
        }

        // ---- O^T += V^T * P^T : 4 d-tiles; V^T A-fragment shared by both q-tiles
        // A element j -> key = j + 16*hf; row m = d = 16t + l15.
        const int kbase = 16 * hf;
        for (int t = 0; t < 4; ++t) {
            const int dcol = 16 * t + l15;
            v16h av;
            for (int i = 0; i < 8; ++i) {
                av[i]     = (_Float16)Vf[cur][(kbase + i) * D_ + dcol];
                av[8 + i] = (_Float16)Vf[cur][(kbase + 8 + i) * D_ + dcol];
            }
            for (int qt = 0; qt < QT; ++qt)
                o[qt][t] = __builtin_amdgcn_wmma_f32_16x16x32_f16(false, av, false, bp[qt],
                                                                  (short)0, o[qt][t], false, false);
        }
    }

    // ---- normalize and store: o[qt][t][r] -> O[b][qg][h][16t + 8*hf + r] ----
    for (int qt = 0; qt < QT; ++qt) {
        const int qg = qbase + qt * 16 + l15;
        const float inv = 1.0f / l_run[qt];
        float* op = O + (((size_t)b * L_ + qg) * H_ + h) * D_;
        for (int t = 0; t < 4; ++t) {
            const int d0 = 16 * t + 8 * hf;
            v4f lo, hi;
            for (int i = 0; i < 4; ++i) { lo[i] = o[qt][t][i] * inv; hi[i] = o[qt][t][4 + i] * inv; }
            *reinterpret_cast<v4f*>(op + d0)     = lo;
            *reinterpret_cast<v4f*>(op + d0 + 4) = hi;
        }
    }
}

extern "C" void kernel_launch(void* const* d_in, const int* in_sizes, int n_in,
                              void* d_out, int out_size, void* d_ws, size_t ws_size,
                              hipStream_t stream) {
    (void)in_sizes; (void)n_in; (void)out_size; (void)d_ws; (void)ws_size;
    const float* Q = (const float*)d_in[0];
    const float* K = (const float*)d_in[1];
    const float* V = (const float*)d_in[2];
    float* O = (float*)d_out;
    dim3 grid(L_ / (64 * QT), H_, B_);   // (query blocks of 128, heads, batch)
    dim3 block(32 * NWAVES);
    fa_fwd_kernel<<<grid, block, 0, stream>>>(Q, K, V, O);
}
